// GraphTransformer_50714973831897
// MI455X (gfx1250) — compile-verified, matrix operand order checked by reading, not run on hardware
//
#include <hip/hip_runtime.h>
#include <hip/hip_bf16.h>
#include <cmath>

// ---------------------------------------------------------------------------
// Problem constants (from reference)
// ---------------------------------------------------------------------------
#define N_NODES 100000
#define N_EDGES 400000
#define N_GRAPH 2048
#define NTOT    (N_NODES + N_GRAPH)            // 102048
#define EBASE   (N_EDGES + 2 * N_NODES)        // 600000 (real + vnode edges)
#define EAUG    (EBASE + NTOT)                 // 702048 (+ self loops)
#define DIM     64
#define HEADS   2
#define LAYERS  3
#define SLOPE   0.01f
#define EPS_LN  1e-5f
#define EPS_GEN 1e-7f

typedef __attribute__((ext_vector_type(16))) __bf16 v16bf;
typedef __attribute__((ext_vector_type(8)))  float  v8f;
typedef __attribute__((ext_vector_type(4)))  unsigned int u32x4;
typedef __attribute__((ext_vector_type(8)))  int i32x8;
typedef __attribute__((ext_vector_type(4)))  int i32x4;

#if defined(__HIP_DEVICE_COMPILE__) && defined(__gfx1250__) && \
    __has_builtin(__builtin_amdgcn_tensor_load_to_lds) && \
    __has_builtin(__builtin_amdgcn_s_wait_tensorcnt)
#define USE_TDM 1
#else
#define USE_TDM 0
#endif

// ---------------------------------------------------------------------------
// WMMA GEMM: C[rows,NOUT](ldc,coff) = act(A[rows,NIN] @ W[NIN,NOUT] + b) (+=C)
// Block = 256 threads = 8 waves; each wave owns a 16-row strip.
// Weight panel is staged per 32-row K-chunk into LDS by the Tensor Data Mover
// (TENSOR_LOAD_TO_LDS, TENSORcnt-synced), then swizzled f32->bf16 into exact
// v_wmma B-fragment layout.
// ---------------------------------------------------------------------------
template <int NIN, int NOUT, int ACT, bool BIAS, bool ACC>
__global__ __launch_bounds__(256)
void k_gemm(const float* __restrict__ A, const float* __restrict__ W,
            const float* __restrict__ bias, float* __restrict__ C,
            int rows, int ldc, int coff)
{
    constexpr int KC = (NIN + 31) / 32;   // K chunks of 32
    constexpr int CT = NOUT / 16;         // column tiles of 16
    __shared__ float  sRaw[32 * NOUT];    // raw f32 weight chunk (TDM target)
    __shared__ __bf16 sW[CT * KC * 512];  // 512 = 32 lanes * 16 bf16

    const int  tid  = threadIdx.x;
    const int  lane = tid & 31;
    const int  wave = tid >> 5;
    const int  kh   = lane >> 4;
    const long row0 = (long)blockIdx.x * 128 + wave * 16;
    const bool active = row0 < rows;

    // ---- A fragments (16x32 bf16 layout), loaded before staging (overlap) --
    v16bf afrag[KC];
    if (active) {
        long rowA = row0 + (lane & 15);
        if (rowA >= rows) rowA = rows - 1;     // clamp; tail masked on store
        const float* arow = A + rowA * NIN;
        for (int kc = 0; kc < KC; ++kc) {
            if constexpr (NIN % 32 == 0) {
                const float* ap = arow + kc * 32 + kh * 8;
                float t[16];
                *(float4*)(t)      = *(const float4*)(ap);
                *(float4*)(t + 4)  = *(const float4*)(ap + 4);
                *(float4*)(t + 8)  = *(const float4*)(ap + 16);
                *(float4*)(t + 12) = *(const float4*)(ap + 20);
                for (int j = 0; j < 16; ++j) afrag[kc][j] = (__bf16)t[j];
            } else {
                for (int j = 0; j < 16; ++j) {
                    int k = kc * 32 + ((j < 8) ? (kh * 8 + j) : (16 + kh * 8 + (j - 8)));
                    afrag[kc][j] = (__bf16)((k < NIN) ? arow[k] : 0.0f);
                }
            }
        }
    }

    // ---- stage weights chunk-by-chunk: TDM -> sRaw -> bf16 fragments in sW --
    for (int kc = 0; kc < KC; ++kc) {
        const int rbase = kc * 32;
        const int rcnt  = (NIN - rbase < 32) ? (NIN - rbase) : 32;
#if USE_TDM
        if (wave == 0) {
            // Tensor DMA descriptor (CDNA5 ISA 8.3/8.4): 2D tile, 4B elems
            unsigned long long ga = (unsigned long long)(const void*)(W + (size_t)rbase * NOUT);
            unsigned ldsOff = (unsigned)(unsigned long long)(const void*)&sRaw[0];
            u32x4 g0;
            g0.x = 1u;                                             // count=1 (user D#)
            g0.y = ldsOff;                                         // lds_addr
            g0.z = (unsigned)(ga & 0xFFFFFFFFull);                 // global_addr lo
            g0.w = (unsigned)((ga >> 32) & 0x01FFFFFFull)          // global_addr hi
                 | (2u << 30);                                     // type=2 (image)
            i32x8 g1;
            g1[0] = (int)(2u << 16);                               // data_size=4B
            g1[1] = (int)(((unsigned)NOUT & 0xFFFFu) << 16);       // tensor_dim0 lo
            g1[2] = (int)((((unsigned)NOUT >> 16) & 0xFFFFu)       // tensor_dim0 hi
                 | (((unsigned)rcnt & 0xFFFFu) << 16));            // tensor_dim1 lo
            g1[3] = (int)(((unsigned)NOUT & 0xFFFFu) << 16);       // tile_dim0
            g1[4] = (int)((unsigned)rcnt & 0xFFFFu);               // tile_dim1
            g1[5] = (int)(unsigned)NOUT;                           // dim0_stride lo
            g1[6] = 0;
            g1[7] = 0;
            i32x4 z4 = (i32x4)0;
#if __clang_major__ >= 23
            i32x8 z8 = (i32x8)0;
            __builtin_amdgcn_tensor_load_to_lds(g0, g1, z4, z4, z8, 0);
#else
            __builtin_amdgcn_tensor_load_to_lds(g0, g1, z4, z4, 0);
#endif
            __builtin_amdgcn_s_wait_tensorcnt(0);
        }
#else
        for (int idx = tid * 4; idx < rcnt * NOUT; idx += 1024)
            *(float4*)&sRaw[idx] = *(const float4*)&W[(size_t)rbase * NOUT + idx];
#endif
        __syncthreads();
        // swizzle chunk into B-fragment layout
        for (int idx = tid; idx < CT * 512; idx += 256) {
            int j   = idx & 15;
            int ln  = (idx >> 4) & 31;
            int ct  = idx >> 9;
            int n   = ct * 16 + (ln & 15);
            int kh2 = ln >> 4;
            int lk  = (j < 8) ? (kh2 * 8 + j) : (16 + kh2 * 8 + (j - 8));
            float w = (lk < rcnt) ? sRaw[lk * NOUT + n] : 0.0f;
            sW[(ct * KC + kc) * 512 + (idx & 511)] = (__bf16)w;
        }
        __syncthreads();   // protect sRaw before next chunk's DMA
    }

    if (!active) return;

    // ---- WMMA compute + store ---------------------------------------------
    for (int ct = 0; ct < CT; ++ct) {
        v8f acc = {};
        if constexpr (BIAS) {
            float bv = bias[ct * 16 + (lane & 15)];
            for (int r = 0; r < 8; ++r) acc[r] = bv;
        }
        for (int kc = 0; kc < KC; ++kc) {
            v16bf bfrag = *(const v16bf*)&sW[(ct * KC + kc) * 512 + lane * 16];
            acc = __builtin_amdgcn_wmma_f32_16x16x32_bf16(
                false, afrag[kc], false, bfrag, (short)0, acc, false, false);
        }
        // C/D layout: vgpr r -> M = r + (lane>=16)*8, N = lane&15
        const int n  = lane & 15;
        const int mb = kh * 8;
        float* cp = C + (row0 + mb) * (long)ldc + (coff + ct * 16 + n);
        if (row0 + 16 <= rows) {           // wave-uniform fast path
#pragma unroll
            for (int r = 0; r < 8; ++r) {
                float v = acc[r];
                if constexpr (ACT == 1) v = (v > 0.0f) ? v : v * SLOPE;
                if constexpr (ACC) *cp += v; else *cp = v;
                cp += ldc;
            }
        } else {
#pragma unroll
            for (int r = 0; r < 8; ++r) {
                if (row0 + mb + r < rows) {
                    float v = acc[r];
                    if constexpr (ACT == 1) v = (v > 0.0f) ? v : v * SLOPE;
                    if constexpr (ACC) *cp += v; else *cp = v;
                }
                cp += ldc;
            }
        }
    }
}

// ---------------------------------------------------------------------------
// Elementwise / scatter kernels
// ---------------------------------------------------------------------------
__global__ void k_fill(float* __restrict__ p, float v, long n) {
    long i = (long)blockIdx.x * blockDim.x + threadIdx.x;
    if (i < n) p[i] = v;
}

__global__ void k_add_inplace(float* __restrict__ a, const float* __restrict__ b, long n) {
    long i = (long)blockIdx.x * blockDim.x + threadIdx.x;
    if (i < n) a[i] += b[i];
}

__global__ void k_topology(const int* __restrict__ ei, const int* __restrict__ batch,
                           int* __restrict__ src, int* __restrict__ dst) {
    int e = blockIdx.x * blockDim.x + threadIdx.x;
    if (e >= EAUG) return;
    int s, d;
    if (e < N_EDGES)                  { s = ei[e];                 d = ei[N_EDGES + e]; }
    else if (e < N_EDGES + N_NODES)   { int i = e - N_EDGES;       s = i; d = batch[i] + N_NODES; }
    else if (e < EBASE)               { int i = e - N_EDGES - N_NODES; s = batch[i] + N_NODES; d = i; }
    else                              { int i = e - EBASE;         s = i; d = i; }
    src[e] = s; dst[e] = d;
}

__global__ void k_abatch(const int* __restrict__ batch, int* __restrict__ ab) {
    int i = blockIdx.x * blockDim.x + threadIdx.x;
    if (i < NTOT) ab[i] = (i < N_NODES) ? batch[i] : (i - N_NODES);
}

__global__ void k_pad_attr(float* __restrict__ ae) {
    long idx = (long)blockIdx.x * blockDim.x + threadIdx.x;
    if (idx >= (long)2 * N_NODES * DIM) return;
    long r = idx >> 6; int j = (int)(idx & 63);
    ae[((long)N_EDGES + r) * DIM + j] = (j == 0) ? 1.0f : 0.0f;
}

__global__ void k_cnt_edges(const int* __restrict__ dst, float* __restrict__ cnt) {
    int e = blockIdx.x * blockDim.x + threadIdx.x;
    if (e < EBASE) atomicAdd(&cnt[dst[e]], 1.0f);
}

__global__ void k_loop_sum(const float* __restrict__ ae, const int* __restrict__ dst,
                           float* __restrict__ lsum) {
    long idx = (long)blockIdx.x * blockDim.x + threadIdx.x;
    if (idx >= (long)EBASE * DIM) return;
    long e = idx >> 6; int j = (int)(idx & 63);
    atomicAdd(&lsum[(long)dst[e] * DIM + j], ae[e * DIM + j]);
}

__global__ void k_loop_attr(const float* __restrict__ lsum, const float* __restrict__ cnt,
                            float* __restrict__ ae) {
    long idx = (long)blockIdx.x * blockDim.x + threadIdx.x;
    if (idx >= (long)NTOT * DIM) return;
    long i = idx >> 6;
    float c = cnt[i]; c = (c < 1.0f) ? 1.0f : c;
    ae[((long)EBASE + i) * DIM + (idx & 63)] = lsum[idx] / c;
}

__global__ void k_cnt_abatch(const int* __restrict__ ab, float* __restrict__ cntA) {
    int i = blockIdx.x * blockDim.x + threadIdx.x;
    if (i < NTOT) atomicAdd(&cntA[ab[i]], 1.0f);
}

// --- graph LayerNorm (mode='graph', affine=False) ---
__global__ void k_ln_stats(const float* __restrict__ o, const int* __restrict__ ab,
                           float* __restrict__ gsum, float* __restrict__ gss) {
    int i = blockIdx.x * blockDim.x + threadIdx.x;
    if (i >= NTOT) return;
    const float* r = o + (long)i * DIM;
    float s = 0.f, ss = 0.f;
    for (int j = 0; j < DIM; ++j) { float v = r[j]; s += v; ss += v * v; }
    int g = ab[i];
    atomicAdd(&gsum[g], s);
    atomicAdd(&gss[g], ss);
}

__global__ void k_ln_final(const float* __restrict__ gsum, const float* __restrict__ gss,
                           const float* __restrict__ cntA, float* __restrict__ meanb,
                           float* __restrict__ invb) {
    int g = blockIdx.x * blockDim.x + threadIdx.x;
    if (g >= N_GRAPH) return;
    float c = cntA[g]; c = (c < 1.f) ? 1.f : c;
    float norm = c * (float)DIM;
    float mean = gsum[g] / norm;
    float var  = gss[g] / norm - mean * mean;   // == E[(x-mean)^2] with same norm
    meanb[g] = mean;
    invb[g]  = rsqrtf(var + EPS_LN);
}

__global__ void k_ln_apply(const float* __restrict__ o, const int* __restrict__ ab,
                           const float* __restrict__ meanb, const float* __restrict__ invb,
                           float* __restrict__ on, float* __restrict__ xt) {
    long idx = (long)blockIdx.x * blockDim.x + threadIdx.x;
    if (idx >= (long)NTOT * DIM) return;
    long i = idx >> 6; int j = (int)(idx & 63);
    int g = ab[i];
    float v = (o[idx] - meanb[g]) * invb[g];
    on[idx] = v;
    if (xt) xt[i * 128 + j] = v;
}

// --- GENConv message + aggregate ---
__global__ void k_gen_msg(const float* __restrict__ on, const float* __restrict__ ae,
                          const int* __restrict__ src, const int* __restrict__ dst,
                          float* __restrict__ agg) {
    long idx = (long)blockIdx.x * blockDim.x + threadIdx.x;
    if (idx >= (long)EAUG * DIM) return;
    long e = idx >> 6; int j = (int)(idx & 63);
    float v = on[(long)src[e] * DIM + j] + ae[idx];
    v = ((v > 0.f) ? v : 0.f) + EPS_GEN;
    atomicAdd(&agg[(long)dst[e] * DIM + j], v);
}

// --- edge softmax attention ---
__device__ inline void atomicMaxF(float* addr, float val) {
    if (val >= 0.0f) atomicMax((int*)addr, __float_as_int(val));
    else             atomicMin((unsigned int*)addr, __float_as_uint(val));
}

__global__ void k_alpha(const float* __restrict__ q, const float* __restrict__ k,
                        const float* __restrict__ ee, const int* __restrict__ src,
                        const int* __restrict__ dst, float* __restrict__ alpha,
                        float* __restrict__ amax) {
    long idx = (long)blockIdx.x * blockDim.x + threadIdx.x;
    if (idx >= (long)EAUG * HEADS) return;
    long e = idx >> 1; int h = (int)(idx & 1);
    const float* qp = q  + (long)dst[e] * 128 + h * 64;
    const float* kp = k  + (long)src[e] * 128 + h * 64;
    const float* ep = ee + e * 128 + h * 64;
    float s = 0.f;
    for (int j = 0; j < 64; ++j) s += qp[j] * (kp[j] + ep[j]);
    s *= 0.125f;                                // 1/sqrt(D)
    alpha[idx] = s;
    atomicMaxF(&amax[(long)dst[e] * HEADS + h], s);
}

__global__ void k_exp(const float* __restrict__ amax, const int* __restrict__ dst,
                      float* __restrict__ alpha, float* __restrict__ denom) {
    long idx = (long)blockIdx.x * blockDim.x + threadIdx.x;
    if (idx >= (long)EAUG * HEADS) return;
    long e = idx >> 1; int h = (int)(idx & 1);
    float ex = expf(alpha[idx] - amax[(long)dst[e] * HEADS + h]);
    alpha[idx] = ex;
    atomicAdd(&denom[(long)dst[e] * HEADS + h], ex);
}

__global__ void k_attn_out(const float* __restrict__ v, const float* __restrict__ ee,
                           const float* __restrict__ alpha, const float* __restrict__ denom,
                           const int* __restrict__ src, const int* __restrict__ dst,
                           float* __restrict__ out) {
    long idx = (long)blockIdx.x * blockDim.x + threadIdx.x;
    if (idx >= (long)EAUG * 128) return;
    long e = idx >> 7; int j = (int)(idx & 127); int h = j >> 6;
    float w  = alpha[e * HEADS + h] / denom[(long)dst[e] * HEADS + h];
    float vv = (v[(long)src[e] * 128 + j] + ee[e * 128 + j]) * w;
    atomicAdd(&out[(long)dst[e] * 128 + j], vv);
}

// --- FiLM conditioning + residual ---
__global__ void k_film(float* __restrict__ o, const float* __restrict__ lh,
                       const float* __restrict__ csg, const int* __restrict__ ab) {
    long idx = (long)blockIdx.x * blockDim.x + threadIdx.x;
    if (idx >= (long)NTOT * DIM) return;
    long i = idx >> 6; int j = (int)(idx & 63);
    const float* cs = csg + (long)ab[i] * 128;
    o[idx] = o[idx] + lh[idx] * cs[j] + cs[64 + j];
}

// --- final pooling & output assembly ---
__global__ void k_pool(const float* __restrict__ o, const int* __restrict__ batch,
                       float* __restrict__ pooled, float* __restrict__ cntn) {
    long idx = (long)blockIdx.x * blockDim.x + threadIdx.x;
    if (idx >= (long)N_NODES * DIM) return;
    long i = idx >> 6; int j = (int)(idx & 63);
    int g = batch[i];
    atomicAdd(&pooled[(long)g * DIM + j], o[idx]);
    if (j == 0) atomicAdd(&cntn[g], 1.0f);
}

__global__ void k_write_out(const float* __restrict__ o, const float* __restrict__ pooled,
                            const float* __restrict__ cntn, float* __restrict__ out) {
    long idx = (long)blockIdx.x * blockDim.x + threadIdx.x;
    const long no = (long)N_NODES * DIM;
    if (idx >= no + (long)N_GRAPH * 128) return;
    if (idx < no) { out[idx] = o[idx]; return; }
    long r = idx - no;
    long g = r >> 7; int j = (int)(r & 127);
    if (j < 64) {
        float c = cntn[g]; c = (c < 1.f) ? 1.f : c;
        out[idx] = pooled[g * DIM + j] / c;
    } else {
        out[idx] = o[((long)N_NODES + g) * DIM + (j - 64)];
    }
}

// ---------------------------------------------------------------------------
// Orchestration
// ---------------------------------------------------------------------------
#define EW(kern, n, ...) kern<<<dim3((unsigned)(((long)(n) + 255) / 256)), dim3(256), 0, stream>>>(__VA_ARGS__)
#define GEMM_GRID(rows) dim3((unsigned)(((rows) + 127) / 128))

extern "C" void kernel_launch(void* const* d_in, const int* in_sizes, int n_in,
                              void* d_out, int out_size, void* d_ws, size_t ws_size,
                              hipStream_t stream) {
    const float* x        = (const float*)d_in[0];
    const float* eattr    = (const float*)d_in[1];
    const float* cond     = (const float*)d_in[2];
    const float* x2h_W    = (const float*)d_in[3];
    const float* x2h_b    = (const float*)d_in[4];
    const float* e2h_W0   = (const float*)d_in[5];
    const float* e2h_b0   = (const float*)d_in[6];
    const float* e2h_W    = (const float*)d_in[7];
    const float* e2h_b    = (const float*)d_in[8];
    const float* c2h_W    = (const float*)d_in[9];
    const float* c2h_b    = (const float*)d_in[10];
    const float* gen_W    = (const float*)d_in[11];
    const float* gen_b    = (const float*)d_in[12];
    const float* q_W      = (const float*)d_in[13];
    const float* q_b      = (const float*)d_in[14];
    const float* k_W      = (const float*)d_in[15];
    const float* k_b      = (const float*)d_in[16];
    const float* v_W      = (const float*)d_in[17];
    const float* v_b      = (const float*)d_in[18];
    const float* e_W      = (const float*)d_in[19];
    const float* skip_W   = (const float*)d_in[20];
    const float* skip_b   = (const float*)d_in[21];
    const float* lin_W    = (const float*)d_in[22];
    const float* lin_b    = (const float*)d_in[23];
    const float* ff_W1    = (const float*)d_in[24];
    const float* ff_b1    = (const float*)d_in[25];
    const float* ff_W2    = (const float*)d_in[26];
    const float* ff_b2    = (const float*)d_in[27];
    const float* cs_W     = (const float*)d_in[28];
    const float* cs_b     = (const float*)d_in[29];
    const int*   batch    = (const int*)d_in[30];
    const int*   eindex   = (const int*)d_in[31];

    // ---- workspace carve (~920 MB; big buffers aliased across phases) ----
    size_t off = 0;
    char* base = (char*)d_ws;
    auto alloc = [&](size_t elems) -> float* {
        float* p = (float*)(base + off);
        off = (off + elems * 4 + 255) & ~(size_t)255;
        return p;
    };
    float* o      = alloc((size_t)NTOT * DIM);
    float* on     = alloc((size_t)NTOT * DIM);
    float* xt     = alloc((size_t)NTOT * 128);
    float* qb     = alloc((size_t)NTOT * 128);
    float* kb     = alloc((size_t)NTOT * 128);
    float* vb     = alloc((size_t)NTOT * 128);
    float* outb   = alloc((size_t)NTOT * 128);
    float* agg    = alloc((size_t)NTOT * DIM);
    float* lhb    = alloc((size_t)NTOT * DIM);
    float* ae     = alloc((size_t)EAUG * DIM);
    float* eeb    = alloc((size_t)EAUG * 128);   // aliases: edge-MLP temps, FF hidden
    float* alphab = alloc((size_t)EAUG * HEADS);
    float* amaxb  = alloc((size_t)NTOT * HEADS);
    float* denomb = alloc((size_t)NTOT * HEADS);
    float* cbuf   = alloc((size_t)N_GRAPH * DIM);
    float* csg    = alloc((size_t)N_GRAPH * 128);
    float* gsum   = alloc(N_GRAPH);
    float* gss    = alloc(N_GRAPH);
    float* meanb  = alloc(N_GRAPH);
    float* invb   = alloc(N_GRAPH);
    float* cntA   = alloc(N_GRAPH);
    float* pooled = alloc((size_t)N_GRAPH * DIM);
    float* cntn   = alloc(N_GRAPH);
    int*   srcb   = (int*)alloc(EAUG);
    int*   dstb   = (int*)alloc(EAUG);
    int*   abatch = (int*)alloc(NTOT);
    // setup-phase aliases (dead buffers reused)
    float* t0   = qb;                       // node MLP temps
    float* t1   = kb;
    float* tg0  = outb;                     // cond MLP temps
    float* tg1  = agg;
    float* te0  = eeb;                      // edge MLP temps
    float* te1  = eeb + (size_t)N_EDGES * DIM;
    float* lsum = qb;                       // self-loop mean accumulators
    float* cntE = vb;
    float* h1   = eeb;                      // FF hidden (ee is dead by FF time)
    (void)ws_size; (void)in_sizes; (void)n_in; (void)out_size;

    // ---- setup: input MLPs ----
    k_gemm<64, 64, 1, true, false><<<GEMM_GRID(N_NODES), 256, 0, stream>>>(x,  x2h_W,            x2h_b,       t0, N_NODES, 64, 0);
    k_gemm<64, 64, 1, true, false><<<GEMM_GRID(N_NODES), 256, 0, stream>>>(t0, x2h_W + 4096,     x2h_b + 64,  t1, N_NODES, 64, 0);
    k_gemm<64, 64, 0, true, false><<<GEMM_GRID(N_NODES), 256, 0, stream>>>(t1, x2h_W + 8192,     x2h_b + 128, o,  N_NODES, 64, 0);

    k_gemm<64, 64, 1, true, false><<<GEMM_GRID(N_GRAPH), 256, 0, stream>>>(cond, c2h_W,          c2h_b,       tg0, N_GRAPH, 64, 0);
    k_gemm<64, 64, 1, true, false><<<GEMM_GRID(N_GRAPH), 256, 0, stream>>>(tg0,  c2h_W + 4096,   c2h_b + 64,  tg1, N_GRAPH, 64, 0);
    k_gemm<64, 64, 0, true, false><<<GEMM_GRID(N_GRAPH), 256, 0, stream>>>(tg1,  c2h_W + 8192,   c2h_b + 128, cbuf, N_GRAPH, 64, 0);
    hipMemcpyAsync(o + (size_t)N_NODES * DIM, cbuf, (size_t)N_GRAPH * DIM * 4,
                   hipMemcpyDeviceToDevice, stream);

    k_gemm<16, 64, 1, true, false><<<GEMM_GRID(N_EDGES), 256, 0, stream>>>(eattr, e2h_W0,        e2h_b0,      te0, N_EDGES, 64, 0);
    k_gemm<64, 64, 1, true, false><<<GEMM_GRID(N_EDGES), 256, 0, stream>>>(te0,   e2h_W,         e2h_b,       te1, N_EDGES, 64, 0);
    k_gemm<64, 64, 0, true, false><<<GEMM_GRID(N_EDGES), 256, 0, stream>>>(te1,   e2h_W + 4096,  e2h_b + 64,  ae,  N_EDGES, 64, 0);

    // ---- setup: augmented graph (vnode edges + mean self-loop attrs) ----
    EW(k_topology, EAUG, eindex, batch, srcb, dstb);
    EW(k_abatch, NTOT, batch, abatch);
    EW(k_pad_attr, (long)2 * N_NODES * DIM, ae);
    EW(k_fill, NTOT, cntE, 0.0f, (long)NTOT);
    EW(k_fill, (long)NTOT * DIM, lsum, 0.0f, (long)NTOT * DIM);
    EW(k_cnt_edges, EBASE, dstb, cntE);
    EW(k_loop_sum, (long)EBASE * DIM, ae, dstb, lsum);
    EW(k_loop_attr, (long)NTOT * DIM, lsum, cntE, ae);
    EW(k_fill, N_GRAPH, cntA, 0.0f, (long)N_GRAPH);
    EW(k_cnt_abatch, NTOT, abatch, cntA);

    const float minf = -INFINITY;

    // ---- layers ----
    for (int l = 0; l < LAYERS; ++l) {
        // FiLM params from (constant) cond embedding
        k_gemm<64, 128, 0, true, false><<<GEMM_GRID(N_GRAPH), 256, 0, stream>>>(
            cbuf, cs_W + (size_t)l * 8192, cs_b + (size_t)l * 128, csg, N_GRAPH, 128, 0);

        // pre-LN -> on, and into xt[:, :64]
        EW(k_fill, N_GRAPH, gsum, 0.0f, (long)N_GRAPH);
        EW(k_fill, N_GRAPH, gss, 0.0f, (long)N_GRAPH);
        EW(k_ln_stats, NTOT, o, abatch, gsum, gss);
        EW(k_ln_final, N_GRAPH, gsum, gss, cntA, meanb, invb);
        EW(k_ln_apply, (long)NTOT * DIM, o, abatch, meanb, invb, on, xt);

        // GENConv: agg = segsum(relu(on[src]+ae)+eps); gen_out -> xt[:, 64:]
        EW(k_fill, (long)NTOT * DIM, agg, 0.0f, (long)NTOT * DIM);
        EW(k_gen_msg, (long)EAUG * DIM, on, ae, srcb, dstb, agg);
        EW(k_add_inplace, (long)NTOT * DIM, agg, on, (long)NTOT * DIM);
        k_gemm<64, 64, 0, true, false><<<GEMM_GRID(NTOT), 256, 0, stream>>>(
            agg, gen_W + (size_t)l * 4096, gen_b + (size_t)l * 64, xt, NTOT, 128, 64);

        // TransformerConv projections
        k_gemm<128, 128, 0, true, false><<<GEMM_GRID(NTOT), 256, 0, stream>>>(
            xt, q_W + (size_t)l * 16384, q_b + (size_t)l * 128, qb, NTOT, 128, 0);
        k_gemm<128, 128, 0, true, false><<<GEMM_GRID(NTOT), 256, 0, stream>>>(
            xt, k_W + (size_t)l * 16384, k_b + (size_t)l * 128, kb, NTOT, 128, 0);
        k_gemm<128, 128, 0, true, false><<<GEMM_GRID(NTOT), 256, 0, stream>>>(
            xt, v_W + (size_t)l * 16384, v_b + (size_t)l * 128, vb, NTOT, 128, 0);
        // skip output initializes attention accumulator (out = attn + skip)
        k_gemm<128, 128, 0, true, false><<<GEMM_GRID(NTOT), 256, 0, stream>>>(
            xt, skip_W + (size_t)l * 16384, skip_b + (size_t)l * 128, outb, NTOT, 128, 0);
        // per-edge attr projection ee
        k_gemm<64, 128, 0, false, false><<<GEMM_GRID(EAUG), 256, 0, stream>>>(
            ae, e_W + (size_t)l * 8192, (const float*)nullptr, eeb, EAUG, 128, 0);

        // edge softmax
        EW(k_fill, (long)NTOT * HEADS, amaxb, minf, (long)NTOT * HEADS);
        EW(k_fill, (long)NTOT * HEADS, denomb, 0.0f, (long)NTOT * HEADS);
        EW(k_alpha, (long)EAUG * HEADS, qb, kb, eeb, srcb, dstb, alphab, amaxb);
        EW(k_exp, (long)EAUG * HEADS, amaxb, dstb, alphab, denomb);
        EW(k_attn_out, (long)EAUG * 128, vb, eeb, alphab, denomb, srcb, dstb, outb);

        // lin + FiLM + residual
        k_gemm<128, 64, 0, true, false><<<GEMM_GRID(NTOT), 256, 0, stream>>>(
            outb, lin_W + (size_t)l * 8192, lin_b + (size_t)l * 64, lhb, NTOT, 64, 0);
        EW(k_film, (long)NTOT * DIM, o, lhb, csg, abatch);

        // FF block with pre-LN, accumulate into o
        EW(k_fill, N_GRAPH, gsum, 0.0f, (long)N_GRAPH);
        EW(k_fill, N_GRAPH, gss, 0.0f, (long)N_GRAPH);
        EW(k_ln_stats, NTOT, o, abatch, gsum, gss);
        EW(k_ln_final, N_GRAPH, gsum, gss, cntA, meanb, invb);
        EW(k_ln_apply, (long)NTOT * DIM, o, abatch, meanb, invb, on, (float*)nullptr);
        k_gemm<64, 256, 1, true, false><<<GEMM_GRID(NTOT), 256, 0, stream>>>(
            on, ff_W1 + (size_t)l * 16384, ff_b1 + (size_t)l * 256, h1, NTOT, 256, 0);
        k_gemm<256, 64, 0, true, true><<<GEMM_GRID(NTOT), 256, 0, stream>>>(
            h1, ff_W2 + (size_t)l * 16384, ff_b2 + (size_t)l * 64, o, NTOT, 64, 0);
    }

    // ---- final pooling and output ----
    EW(k_fill, (long)N_GRAPH * DIM, pooled, 0.0f, (long)N_GRAPH * DIM);
    EW(k_fill, N_GRAPH, cntn, 0.0f, (long)N_GRAPH);
    EW(k_pool, (long)N_NODES * DIM, o, batch, pooled, cntn);
    EW(k_write_out, (long)N_NODES * DIM + (long)N_GRAPH * 128, o, pooled, cntn, (float*)d_out);
}
